// GraphEncoder_81518479278100
// MI455X (gfx1250) — compile-verified, matrix operand order checked by reading, not run on hardware
//
#include <hip/hip_runtime.h>
#include <hip/hip_bf16.h>

// ---------------------------------------------------------------------------
// GraphEncoder for MI455X (gfx1250, wave32, WMMA).
// GCN edge-scatter -> dense per-graph 128x128 adjacency GEMMs.
// All matmuls via v_wmma_f32_16x16x32_bf16 (fp32 accumulate).
// B operands stored "fragment-ready": per (32-K-block, column) 64B chunk, so
// every fragment (A and B) is built from two 16-byte vector loads.
// ---------------------------------------------------------------------------

typedef __attribute__((ext_vector_type(16))) __bf16 v16bf;
typedef __attribute__((ext_vector_type(8)))  __bf16 v8bf;
typedef __attribute__((ext_vector_type(8)))  float  v8f;

#define NPG   120
#define FEAT  256
#define APAD  128   // padded per-graph adjacency dim

__device__ __forceinline__ __bf16 f2bf(float x) { return (__bf16)x; }
__device__ __forceinline__ float  bf2f(__bf16 x) { return (float)x; }

// Swizzled-B addressing: chunk base for K-block (k0/32) and column n.
__device__ __forceinline__ size_t bsw_base(int k0, int n, int Nc) {
    return ((size_t)(k0 >> 5) * Nc + n) * 32;
}

// ---------------------------------------------------------------------------
// bf16 GEMM: C = [relu]( A[M,K] @ B[K,Nc] + bias ), A row-major, B swizzled.
// Wave -> 64x16 C block (4 stacked 16x16 tiles share one B fragment).
// 8 waves/block -> 128 columns. grid.x = M/64, grid.y = Nc/128, block = 256.
// OUT_MODE: 0 = bf16 row-major, 1 = fp32 row-major, 2 = bf16 Y-swizzle.
// ---------------------------------------------------------------------------
template <bool RELU, bool HAS_BIAS, int OUT_MODE>
__global__ void gemm_bf16_kernel(const __bf16* __restrict__ A,
                                 const __bf16* __restrict__ Bs,
                                 const float*  __restrict__ bias,
                                 __bf16* __restrict__ Cb,
                                 float*  __restrict__ Cf,
                                 int K, int lda, int Nc, int ldc) {
    const int lane  = threadIdx.x & 31;
    const int wave  = threadIdx.x >> 5;
    const int mbase = blockIdx.x * 64;
    const int n     = blockIdx.y * 128 + wave * 16 + (lane & 15);
    const int marow = mbase + (lane & 15);               // A-fragment base row
    const int kb    = (lane >> 4) << 3;                  // K-phase: 0 or 8

    v8f acc0 = {}, acc1 = {}, acc2 = {}, acc3 = {};
    for (int k0 = 0; k0 < K; k0 += 32) {
        const __bf16* bp = Bs + bsw_base(k0, n, Nc) + kb;
        v8bf blo = *reinterpret_cast<const v8bf*>(bp);
        v8bf bhi = *reinterpret_cast<const v8bf*>(bp + 16);
        v16bf b;
#pragma unroll
        for (int i = 0; i < 8; ++i) { b[i] = blo[i]; b[8 + i] = bhi[i]; }
#pragma unroll
        for (int t = 0; t < 4; ++t) {
            const __bf16* ap = A + (size_t)(marow + t * 16) * lda + k0 + kb;
            v8bf alo = *reinterpret_cast<const v8bf*>(ap);       // K = kb..kb+7
            v8bf ahi = *reinterpret_cast<const v8bf*>(ap + 16);  // K = kb+16..+23
            v16bf a;
#pragma unroll
            for (int i = 0; i < 8; ++i) { a[i] = alo[i]; a[8 + i] = ahi[i]; }
            v8f& acc = (t == 0) ? acc0 : (t == 1) ? acc1 : (t == 2) ? acc2 : acc3;
            acc = __builtin_amdgcn_wmma_f32_16x16x32_bf16(false, a, false, b,
                                                          (short)0, acc, false, false);
        }
    }
    const float bv = HAS_BIAS ? bias[n] : 0.0f;
    const int rb = mbase + ((lane >> 4) << 3);           // C: vgpr v -> row v (+8 hi half)
#pragma unroll
    for (int t = 0; t < 4; ++t) {
        v8f& acc = (t == 0) ? acc0 : (t == 1) ? acc1 : (t == 2) ? acc2 : acc3;
#pragma unroll
        for (int v = 0; v < 8; ++v) {
            unsigned row = rb + t * 16 + v;
            float x = acc[v] + bv;
            if (RELU) x = fmaxf(x, 0.0f);
            if (OUT_MODE == 1) {
                Cf[(size_t)row * ldc + n] = x;
            } else if (OUT_MODE == 0) {
                Cb[(size_t)row * ldc + n] = f2bf(x);
            } else {
                // Y-swizzle: row -> (graph g, local row lr); chunk = lr/32.
                unsigned g  = row / NPG;
                unsigned lr = row - g * NPG;
                size_t addr = (((size_t)g * 4 + (lr >> 5)) * FEAT + n) * 32 + (lr & 31);
                Cb[addr] = f2bf(x);
            }
        }
    }
}

// ---------------------------------------------------------------------------
// Per-graph aggregation: H[g,m,:] = relu( sum_k A[g,m,k] * Y[g,k,:] + bias ).
// A bf16 [G][128][128] row-major (zero padded past 120), Y swizzled bf16.
// Wave -> 64x16 block. blockIdx.x = g*4 + (rowquad<<1 | colhalf), block = 256.
// ---------------------------------------------------------------------------
__global__ void graph_agg_kernel(const __bf16* __restrict__ Ab,
                                 const __bf16* __restrict__ Ys,
                                 const float* __restrict__ bias,
                                 __bf16* __restrict__ H) {
    const int lane = threadIdx.x & 31;
    const int wave = threadIdx.x >> 5;
    const int g    = blockIdx.x >> 2;
    const int rq   = (blockIdx.x >> 1) & 1;              // row quad: rows rq*64..+63
    const int ch   = blockIdx.x & 1;                     // col half: cols ch*128..+127
    const int n    = ch * 128 + wave * 16 + (lane & 15);
    const int arow = rq * 64 + (lane & 15);
    const int kb   = (lane >> 4) << 3;
    const __bf16* Ag = Ab + (size_t)g * APAD * APAD;
    const __bf16* Yg = Ys + (size_t)g * 4 * FEAT * 32;

    v8f acc0 = {}, acc1 = {}, acc2 = {}, acc3 = {};
#pragma unroll
    for (int k0 = 0; k0 < APAD; k0 += 32) {
        const __bf16* bp = Yg + ((size_t)(k0 >> 5) * FEAT + n) * 32 + kb;
        v8bf blo = *reinterpret_cast<const v8bf*>(bp);
        v8bf bhi = *reinterpret_cast<const v8bf*>(bp + 16);
        v16bf b;
#pragma unroll
        for (int i = 0; i < 8; ++i) { b[i] = blo[i]; b[8 + i] = bhi[i]; }
#pragma unroll
        for (int t = 0; t < 4; ++t) {
            const __bf16* ap = Ag + (size_t)(arow + t * 16) * APAD + k0 + kb;
            v8bf alo = *reinterpret_cast<const v8bf*>(ap);
            v8bf ahi = *reinterpret_cast<const v8bf*>(ap + 16);
            v16bf a;
#pragma unroll
            for (int i = 0; i < 8; ++i) { a[i] = alo[i]; a[8 + i] = ahi[i]; }
            v8f& acc = (t == 0) ? acc0 : (t == 1) ? acc1 : (t == 2) ? acc2 : acc3;
            acc = __builtin_amdgcn_wmma_f32_16x16x32_bf16(false, a, false, b,
                                                          (short)0, acc, false, false);
        }
    }
    const float bv = bias[n];
    const int rb = rq * 64 + ((lane >> 4) << 3);
#pragma unroll
    for (int t = 0; t < 4; ++t) {
        v8f& acc = (t == 0) ? acc0 : (t == 1) ? acc1 : (t == 2) ? acc2 : acc3;
#pragma unroll
        for (int v = 0; v < 8; ++v) {
            int rl = rb + t * 16 + v;
            if (rl < NPG) {
                float x = fmaxf(acc[v] + bv, 0.0f);
                H[((size_t)g * NPG + rl) * FEAT + n] = f2bf(x);
            }
        }
    }
}

// ---------------------------------------------------------------------------
// Small helper kernels
// ---------------------------------------------------------------------------
__global__ void cvt_f32_bf16_kernel(const float* __restrict__ s, __bf16* __restrict__ d, int n) {
    int i = blockIdx.x * blockDim.x + threadIdx.x;
    if (i < n) d[i] = f2bf(s[i]);
}

// fp32 row-major W[K][Nc] -> bf16 fragment-ready swizzle.
__global__ void swz_w_kernel(const float* __restrict__ W, __bf16* __restrict__ Bs,
                             int K, int Nc) {
    int i = blockIdx.x * blockDim.x + threadIdx.x;
    if (i < K * Nc) {
        int k = i / Nc, n = i - k * Nc;
        Bs[bsw_base(k & ~31, n, Nc) + (k & 31)] = f2bf(W[i]);
    }
}

// M01[e][j] = b_exist[j] + sum_k relu(emb[e][k]) * W_exist[k][j]   (1 block x 256)
__global__ void m01_kernel(const float* __restrict__ emb, const float* __restrict__ We,
                           const float* __restrict__ be, float* __restrict__ M01) {
    int j = threadIdx.x;
    for (int e = 0; e < 2; ++e) {
        float acc = be[j];
        for (int k = 0; k < 128; ++k)
            acc += fmaxf(emb[e * 128 + k], 0.0f) * We[k * FEAT + j];
        M01[e * FEAT + j] = acc;
    }
}

// X0[n] = [ relu(pos@Wp+bp) | relu(size@Ws+bs) | relu(M01[exist] + We[128+order]) ]
__global__ void x0_kernel(const float* __restrict__ pos, const float* __restrict__ siz,
                          const int* __restrict__ exist,
                          const float* __restrict__ Wp, const float* __restrict__ bp,
                          const float* __restrict__ Ws, const float* __restrict__ bs,
                          const float* __restrict__ We, const float* __restrict__ M01,
                          __bf16* __restrict__ X0) {
    const int n = blockIdx.x, t = threadIdx.x;
    __bf16* row = X0 + (size_t)n * 512;
    if (t < 128) {
        float p = fmaxf(pos[2 * n] * Wp[t] + pos[2 * n + 1] * Wp[128 + t] + bp[t], 0.0f);
        float s = fmaxf(siz[2 * n] * Ws[t] + siz[2 * n + 1] * Ws[128 + t] + bs[t], 0.0f);
        row[t] = f2bf(p);
        row[128 + t] = f2bf(s);
    }
    const int order = n % NPG;
    float e = fmaxf(M01[exist[n] * FEAT + t] + We[(size_t)(128 + order) * FEAT + t], 0.0f);
    row[256 + t] = f2bf(e);
}

__global__ void deg_scatter_kernel(const int* __restrict__ col, float* __restrict__ deg, int E) {
    int i = blockIdx.x * blockDim.x + threadIdx.x;
    if (i < E) atomicAdd(&deg[col[i]], 1.0f);
}

__global__ void dinv_kernel(float* __restrict__ deg, int N) {   // in-place: deg -> rsqrt(deg+1)
    int i = blockIdx.x * blockDim.x + threadIdx.x;
    if (i < N) deg[i] = rsqrtf(deg[i] + 1.0f);
}

__global__ void adj_edges_kernel(const int* __restrict__ row, const int* __restrict__ col,
                                 const float* __restrict__ dinv, float* __restrict__ A, int E) {
    int i = blockIdx.x * blockDim.x + threadIdx.x;
    if (i < E) {
        int r = row[i], c = col[i];
        int g = c / NPG;
        float nrm = dinv[r] * dinv[c];
        atomicAdd(&A[((size_t)g * APAD + (c % NPG)) * APAD + (r % NPG)], nrm);
    }
}

__global__ void adj_self_kernel(const float* __restrict__ dinv, float* __restrict__ A, int N) {
    int i = blockIdx.x * blockDim.x + threadIdx.x;
    if (i < N) {
        int g = i / NPG, l = i % NPG;
        atomicAdd(&A[((size_t)g * APAD + l) * APAD + l], dinv[i] * dinv[i]);
    }
}

// G[g, slot*256 + j] = max over 120 nodes of H   (grid = G, block = 256)
__global__ void pool_kernel(const __bf16* __restrict__ H, __bf16* __restrict__ G, int slot) {
    int g = blockIdx.x, j = threadIdx.x;
    float m = -3.402823466e38f;
    const __bf16* base = H + (size_t)g * NPG * FEAT + j;
    for (int i = 0; i < NPG; ++i) m = fmaxf(m, bf2f(base[(size_t)i * FEAT]));
    G[(size_t)g * 1280 + slot * FEAT + j] = f2bf(m);
}

// ---------------------------------------------------------------------------
extern "C" void kernel_launch(void* const* d_in, const int* in_sizes, int n_in,
                              void* d_out, int out_size, void* d_ws, size_t ws_size,
                              hipStream_t stream) {
    const float* pos    = (const float*)d_in[0];
    const float* siz    = (const float*)d_in[1];
    const int*   exist  = (const int*)d_in[2];
    const int*   eidx   = (const int*)d_in[4];
    const float* W_pos  = (const float*)d_in[5];
    const float* b_pos  = (const float*)d_in[6];
    const float* W_size = (const float*)d_in[7];
    const float* b_size = (const float*)d_in[8];
    const float* emb    = (const float*)d_in[9];
    const float* W_ex   = (const float*)d_in[10];
    const float* b_ex   = (const float*)d_in[11];
    const float* W_node = (const float*)d_in[12];
    const float* b_node = (const float*)d_in[13];
    const float* conv_W = (const float*)d_in[14];
    const float* conv_b = (const float*)d_in[15];
    const float* W_agg  = (const float*)d_in[16];
    const float* b_agg  = (const float*)d_in[17];
    const float* W_mu   = (const float*)d_in[18];
    const float* b_mu   = (const float*)d_in[19];
    const float* W_var  = (const float*)d_in[20];
    const float* b_var  = (const float*)d_in[21];

    const int N = in_sizes[0] / 2;          // 245760 nodes
    const int E = in_sizes[4] / 2;          // 1048576 edges
    const int G = N / NPG;                  // 2048 graphs
    const int T = 4;
    const int* erow = eidx;
    const int* ecol = eidx + E;

    // ---- workspace carve-up ----
    char* base = (char*)d_ws;
    size_t off = 0;
    auto take = [&](size_t bytes) { size_t o = off; off = (off + bytes + 255) & ~(size_t)255; return (void*)(base + o); };
    __bf16* X0    = (__bf16*)take((size_t)N * 512 * 2);
    __bf16* H0    = (__bf16*)take((size_t)N * FEAT * 2);
    __bf16* H1    = (__bf16*)take((size_t)N * FEAT * 2);
    __bf16* Ysw   = (__bf16*)take((size_t)G * 4 * FEAT * 32 * 2);   // swizzled Y
    float*  Adj   = (float*)take((size_t)G * APAD * APAD * 4);
    __bf16* Adjb  = (__bf16*)take((size_t)G * APAD * APAD * 2);
    float*  deg   = (float*)take((size_t)N * 4);
    float*  M01   = (float*)take(2 * FEAT * 4);
    __bf16* Gmat  = (__bf16*)take((size_t)G * 1280 * 2);
    __bf16* Lat   = (__bf16*)take((size_t)G * FEAT * 2);
    __bf16* Wn_s  = (__bf16*)take((size_t)512 * FEAT * 2);
    __bf16* Wc_s  = (__bf16*)take((size_t)T * FEAT * FEAT * 2);
    __bf16* Wa_s  = (__bf16*)take((size_t)1280 * FEAT * 2);
    __bf16* Wm_s  = (__bf16*)take((size_t)FEAT * FEAT * 2);
    __bf16* Wv_s  = (__bf16*)take((size_t)FEAT * FEAT * 2);

    // ---- init / zero ----
    hipMemsetAsync(deg, 0, (size_t)N * 4, stream);
    hipMemsetAsync(Adj, 0, (size_t)G * APAD * APAD * 4, stream);
    hipMemsetAsync(Ysw, 0, (size_t)G * 4 * FEAT * 32 * 2, stream);  // pad slots stay 0

    // ---- swizzle weights to fragment-ready bf16 ----
    auto swz = [&](const float* s, __bf16* d, int K, int Nc) {
        int n = K * Nc;
        swz_w_kernel<<<(n + 255) / 256, 256, 0, stream>>>(s, d, K, Nc);
    };
    swz(W_node, Wn_s, 512, FEAT);
    for (int t = 0; t < T; ++t)
        swz(conv_W + (size_t)t * FEAT * FEAT, Wc_s + (size_t)t * FEAT * FEAT, FEAT, FEAT);
    swz(W_agg, Wa_s, 1280, FEAT);
    swz(W_mu,  Wm_s, FEAT, FEAT);
    swz(W_var, Wv_s, FEAT, FEAT);

    // ---- front end: exist table, node features, big input GEMM ----
    m01_kernel<<<1, 256, 0, stream>>>(emb, W_ex, b_ex, M01);
    x0_kernel<<<N, 256, 0, stream>>>(pos, siz, exist, W_pos, b_pos, W_size, b_size,
                                     W_ex, M01, X0);
    // x = relu(X0 @ W_node + b_node) -> H0
    gemm_bf16_kernel<true, true, 0><<<dim3(N / 64, FEAT / 128), 256, 0, stream>>>(
        X0, Wn_s, b_node, H0, nullptr, 512, 512, FEAT, FEAT);

    // ---- GCN normalization: dense per-graph adjacency (fp32 build, bf16 copy) ----
    deg_scatter_kernel<<<(E + 255) / 256, 256, 0, stream>>>(ecol, deg, E);
    dinv_kernel<<<(N + 255) / 256, 256, 0, stream>>>(deg, N);
    adj_edges_kernel<<<(E + 255) / 256, 256, 0, stream>>>(erow, ecol, deg, Adj, E);
    adj_self_kernel<<<(N + 255) / 256, 256, 0, stream>>>(deg, Adj, N);
    cvt_f32_bf16_kernel<<<(G * APAD * APAD + 255) / 256, 256, 0, stream>>>(
        Adj, Adjb, G * APAD * APAD);

    // ---- pool(x) into slot 0 ----
    pool_kernel<<<G, 256, 0, stream>>>(H0, Gmat, 0);

    // ---- T GCN layers: Ysw = H @ W_t (swizzled) ; H' = relu(A @ Y + b_t) ; pool ----
    __bf16* Hin = H0;
    __bf16* Hout = H1;
    for (int t = 0; t < T; ++t) {
        gemm_bf16_kernel<false, false, 2><<<dim3(N / 64, FEAT / 128), 256, 0, stream>>>(
            Hin, Wc_s + (size_t)t * FEAT * FEAT, nullptr, Ysw, nullptr,
            FEAT, FEAT, FEAT, 0);
        graph_agg_kernel<<<G * 4, 256, 0, stream>>>(Adjb, Ysw, conv_b + t * FEAT, Hout);
        pool_kernel<<<G, 256, 0, stream>>>(Hout, Gmat, t + 1);
        __bf16* tmp = Hin; Hin = Hout; Hout = tmp;
    }

    // ---- readout: latent = G @ W_agg + b_agg ; mu/log_var ----
    gemm_bf16_kernel<false, true, 0><<<dim3(G / 64, FEAT / 128), 256, 0, stream>>>(
        Gmat, Wa_s, b_agg, Lat, nullptr, 1280, 1280, FEAT, FEAT);
    float* out_mu = (float*)d_out;
    float* out_lv = out_mu + (size_t)G * FEAT;
    gemm_bf16_kernel<false, true, 1><<<dim3(G / 64, FEAT / 128), 256, 0, stream>>>(
        Lat, Wm_s, b_mu, nullptr, out_mu, FEAT, FEAT, FEAT, FEAT);
    gemm_bf16_kernel<false, true, 1><<<dim3(G / 64, FEAT / 128), 256, 0, stream>>>(
        Lat, Wv_s, b_var, nullptr, out_lv, FEAT, FEAT, FEAT, FEAT);
}
